// SNN_1511828488906
// MI455X (gfx1250) — matrix operand order repaired and checked by reading, not compiled, not run on hardware
//
#include <hip/hip_runtime.h>
#include <hip/hip_bf16.h>

// ---------------------------------------------------------------------------
// SNN forward: conv1+IF+pool -> conv2(WMMA)+IF+pool -> fc1(WMMA)+IF -> fc2+IF+mean
// All on-chip intermediates fp16 (spikes are exactly 0/1). fc1 weights (268MB
// fp32) are the bandwidth floor; read once, cvt to fp16 inline.
// ---------------------------------------------------------------------------

#define TT   8
#define NB   32
#define CMID 64

typedef __attribute__((ext_vector_type(16))) _Float16 v16h;
typedef __attribute__((ext_vector_type(8)))  _Float16 v8h;
typedef __attribute__((ext_vector_type(4)))  _Float16 v4h;
typedef __attribute__((ext_vector_type(8)))  float    v8f;

static __device__ __forceinline__ v16h mk16(v8h lo, v8h hi) {
  v16h r;
#pragma unroll
  for (int i = 0; i < 8; ++i) { r[i] = lo[i]; r[i + 8] = hi[i]; }
  return r;
}

static __device__ __forceinline__ v8f wmma16(v16h a, v16h b, v8f c) {
  // D = A(16x32 f16) * B(32x16 f16) + C(16x16 f32)
  return __builtin_amdgcn_wmma_f32_16x16x32_f16(
      false, a, false, b, (short)0, c, false, false);
}

// ---------------------------------------------------------------------------
// Kernel 0: w2 fp32 -> fp16, flat [64][576] (o, c*9+ky*3+kx) == source layout
// ---------------------------------------------------------------------------
__global__ void k_w2h(const float* __restrict__ w2, _Float16* __restrict__ w2h) {
  int i = blockIdx.x * 256 + threadIdx.x;
  if (i < 64 * 576) w2h[i] = (_Float16)w2[i];
}

// ---------------------------------------------------------------------------
// Kernel 1: conv1 (3->64, 3x3 pad1) + IF over T + 2x2 maxpool -> s1 fp16
// block = (n, ty, tx): 8x8 spatial tile of the 64x64 image, all 64 channels,
// membrane v kept in LDS across the t loop.
// ---------------------------------------------------------------------------
__global__ __launch_bounds__(256) void k_conv1(const float* __restrict__ x,
                                               const float* __restrict__ w1,
                                               _Float16* __restrict__ s1) {
  __shared__ float    vmem[64 * 64];       // [ch][pos]
  __shared__ float    patch[3 * 10 * 10];  // input tile + halo
  __shared__ float    wl[64 * 27];
  __shared__ _Float16 sb[64 * 64];         // spikes this step

  const int n  = blockIdx.x;
  const int ty = blockIdx.y;
  const int tx = blockIdx.z;
  const int tid = threadIdx.x;

  for (int i = tid; i < 64 * 27; i += 256) wl[i] = w1[i];
  for (int i = tid; i < 64 * 64; i += 256) vmem[i] = 0.f;
  __syncthreads();

  const int pos = tid & 63, ly = pos >> 3, lx = pos & 7;
  const int c0 = (tid >> 6) * 16;

  for (int t = 0; t < TT; ++t) {
    for (int i = tid; i < 300; i += 256) {
      int c = i / 100, r = i % 100, yy = r / 10, xx = r % 10;
      int gy = ty * 8 + yy - 1, gx = tx * 8 + xx - 1;
      float val = 0.f;
      if (gy >= 0 && gy < 64 && gx >= 0 && gx < 64)
        val = x[(((size_t)(t * NB + n) * 3 + c) << 12) + gy * 64 + gx];
      patch[i] = val;
    }
    __syncthreads();

    float acc[16];
#pragma unroll
    for (int j = 0; j < 16; ++j) acc[j] = 0.f;
    for (int c = 0; c < 3; ++c)
      for (int ky = 0; ky < 3; ++ky)
#pragma unroll
        for (int kx = 0; kx < 3; ++kx) {
          float p = patch[c * 100 + (ly + ky) * 10 + (lx + kx)];
          int widx = c * 9 + ky * 3 + kx;
#pragma unroll
          for (int j = 0; j < 16; ++j)
            acc[j] = fmaf(p, wl[(c0 + j) * 27 + widx], acc[j]);
        }
    // IF: v += x; spike = v>=1; hard reset
#pragma unroll
    for (int j = 0; j < 16; ++j) {
      int idx = (c0 + j) * 64 + pos;
      float vv = vmem[idx] + acc[j];
      float s = (vv >= 1.0f) ? 1.f : 0.f;
      vmem[idx] = (s > 0.f) ? 0.f : vv;
      sb[idx] = (_Float16)s;
    }
    __syncthreads();
    // 2x2 max pool on spikes -> s1[t][n][c][32][32]
    for (int i = tid; i < 1024; i += 256) {
      int c = i >> 4, q = i & 15, py = q >> 2, px = q & 3;
      _Float16 a = sb[c * 64 + (2 * py) * 8 + 2 * px];
      _Float16 b = sb[c * 64 + (2 * py) * 8 + 2 * px + 1];
      _Float16 cc = sb[c * 64 + (2 * py + 1) * 8 + 2 * px];
      _Float16 d = sb[c * 64 + (2 * py + 1) * 8 + 2 * px + 1];
      _Float16 m = a > b ? a : b;
      _Float16 m2 = cc > d ? cc : d;
      m = m > m2 ? m : m2;
      s1[(((size_t)(t * NB + n) * 64 + c) << 10) + (ty * 4 + py) * 32 +
         (tx * 4 + px)] = m;
    }
    __syncthreads();
  }
}

// ---------------------------------------------------------------------------
// Kernel 2: conv2 (64->64) as WMMA GEMM C[64ch,64pos] = W[64,576] x Xcol[576,64]
// per timestep; + IF (v in LDS) + pool -> s2 fp16 [T][N][16384].
// block = (n, ty, tx): 8x8 tile of the 32x32 image. 8 waves: wave w covers
// M-tile w>>1, N-tiles {2*(w&1), 2*(w&1)+1}.
// ---------------------------------------------------------------------------
__global__ __launch_bounds__(256) void k_conv2(const _Float16* __restrict__ s1,
                                               const _Float16* __restrict__ w2h,
                                               _Float16* __restrict__ s2) {
  __shared__ float    vmem[64 * 64];     // 16 KB  [ch][pos]
  __shared__ _Float16 stile[64 * 100];   // 12.8 KB spikes + halo
  __shared__ _Float16 xcol[64 * 576];    // 73.7 KB im2col [pos][k]
  __shared__ _Float16 sb[64 * 64];       // 8 KB spikes

  const int n = blockIdx.x, ty = blockIdx.y, tx = blockIdx.z;
  const int tid = threadIdx.x, lane = tid & 31, wv = tid >> 5;
  const int mt = wv >> 1;
  const int ntb = (wv & 1) * 2;

  for (int i = tid; i < 64 * 64; i += 256) vmem[i] = 0.f;
  __syncthreads();

  for (int t = 0; t < TT; ++t) {
    // stage spike tile with halo
    for (int i = tid; i < 6400; i += 256) {
      int c = i / 100, r = i % 100, yy = r / 10, xx = r % 10;
      int gy = ty * 8 + yy - 1, gx = tx * 8 + xx - 1;
      _Float16 val = (_Float16)0.f;
      if (gy >= 0 && gy < 32 && gx >= 0 && gx < 32)
        val = s1[(((size_t)(t * NB + n) * 64 + c) << 10) + gy * 32 + gx];
      stile[i] = val;
    }
    __syncthreads();
    // im2col: xcol[pos][k], k = c*9 + ky*3 + kx (576 = 18*32 exactly)
    for (int i = tid; i < 64 * 576; i += 256) {
      int p = i / 576, k = i % 576;
      int c = k / 9, r = k % 9, ky = r / 3, kx = r % 3;
      int lly = p >> 3, llx = p & 7;
      xcol[i] = stile[c * 100 + (lly + ky) * 10 + (llx + kx)];
    }
    __syncthreads();

    v8f c0v = {}, c1v = {};
#pragma unroll 2
    for (int ks = 0; ks < 18; ++ks) {
      // A fragment: W row m, K chunks {k0..k0+7, k0+16..k0+23}
      int m = mt * 16 + (lane & 15);
      int k0 = ks * 32 + ((lane >> 4) << 3);
      const v8h* pa = (const v8h*)(w2h + m * 576 + k0);
      v16h a = mk16(pa[0], pa[2]);
      // B fragments: column = pos, K chunks {kb..kb+7, kb+16..kb+23}
      int kb = ks * 32 + ((lane >> 4) << 3);
      const v8h* pb0 = (const v8h*)(xcol + (ntb * 16 + (lane & 15)) * 576 + kb);
      const v8h* pb1 =
          (const v8h*)(xcol + ((ntb + 1) * 16 + (lane & 15)) * 576 + kb);
      v16h b0 = mk16(pb0[0], pb0[2]);
      v16h b1 = mk16(pb1[0], pb1[2]);
      c0v = wmma16(a, b0, c0v);
      c1v = wmma16(a, b1, c1v);
    }
    // IF update: C layout M = r + 8*(lane>>4), N = lane&15
#pragma unroll
    for (int r = 0; r < 8; ++r) {
      int ch = mt * 16 + r + ((lane >> 4) << 3);
      {
        int p = ntb * 16 + (lane & 15);
        int idx = ch * 64 + p;
        float vv = vmem[idx] + c0v[r];
        float s = (vv >= 1.f) ? 1.f : 0.f;
        vmem[idx] = (s > 0.f) ? 0.f : vv;
        sb[idx] = (_Float16)s;
      }
      {
        int p = (ntb + 1) * 16 + (lane & 15);
        int idx = ch * 64 + p;
        float vv = vmem[idx] + c1v[r];
        float s = (vv >= 1.f) ? 1.f : 0.f;
        vmem[idx] = (s > 0.f) ? 0.f : vv;
        sb[idx] = (_Float16)s;
      }
    }
    __syncthreads();
    // pool 2x2 -> s2[t][n][c*256 + y*16 + x] (matches reshape order)
    for (int i = tid; i < 1024; i += 256) {
      int c = i >> 4, q = i & 15, py = q >> 2, px = q & 3;
      _Float16 a = sb[c * 64 + (2 * py) * 8 + 2 * px];
      _Float16 b = sb[c * 64 + (2 * py) * 8 + 2 * px + 1];
      _Float16 cc = sb[c * 64 + (2 * py + 1) * 8 + 2 * px];
      _Float16 d = sb[c * 64 + (2 * py + 1) * 8 + 2 * px + 1];
      _Float16 m = a > b ? a : b;
      _Float16 m2 = cc > d ? cc : d;
      m = m > m2 ? m : m2;
      s2[(size_t)(t * NB + n) * 16384 + c * 256 + (ty * 4 + py) * 16 +
         (tx * 4 + px)] = m;
    }
    __syncthreads();
  }
}

// ---------------------------------------------------------------------------
// Kernel 3: fc1 GEMM [256 rows x 16384] x [16384 x 4096] via WMMA + IF.
// 64 blocks, each owns a disjoint 64-feature strip x all 256 (t,n) rows ->
// fc1 weights (268MB) read exactly once, cvt fp32->fp16 inline.
// 8 waves: wave w -> M-tiles {2w, 2w+1}, 4 N-tiles each.
// ---------------------------------------------------------------------------
__global__ __launch_bounds__(256) void k_fc1(const _Float16* __restrict__ s2,
                                             const float* __restrict__ fc1w,
                                             _Float16* __restrict__ s3) {
  __shared__ _Float16 Al[256 * 32];  // 16 KB activations chunk
  __shared__ _Float16 Bl[64 * 32];   // 4 KB  weights chunk (fp16)
  __shared__ float    hb[256 * 64];  // 64 KB GEMM result for IF scan

  const int fb = blockIdx.x * 64;
  const int tid = threadIdx.x, lane = tid & 31, wv = tid >> 5;
  const int mt0 = wv * 2;

  v8f acc[2][4] = {};

  for (int ks = 0; ks < 512; ++ks) {
    // stage A: 256x32 halfs, 4 x 16B chunks per thread
#pragma unroll
    for (int j = 0; j < 4; ++j) {
      int ch = tid * 4 + j;
      int row = ch >> 2, off = (ch & 3) * 8;
      *(v8h*)(Al + row * 32 + off) =
          *(const v8h*)(s2 + (size_t)row * 16384 + ks * 32 + off);
    }
    // stage B: 64x32 fp32 -> fp16, 2 float4 per thread
#pragma unroll
    for (int j = 0; j < 2; ++j) {
      int q = tid * 2 + j;
      int f = q >> 3, off = (q & 7) * 4;
      float4 w4 = *(const float4*)(fc1w + (size_t)(fb + f) * 16384 + ks * 32 + off);
      v4h h;
      h[0] = (_Float16)w4.x; h[1] = (_Float16)w4.y;
      h[2] = (_Float16)w4.z; h[3] = (_Float16)w4.w;
      *(v4h*)(Bl + f * 32 + off) = h;
    }
    __syncthreads();

    // B fragments shared across both M-tiles
    v16h bfr[4];
#pragma unroll
    for (int nt = 0; nt < 4; ++nt) {
      const v8h* pb =
          (const v8h*)(Bl + (nt * 16 + (lane & 15)) * 32 + ((lane >> 4) << 3));
      bfr[nt] = mk16(pb[0], pb[2]);
    }
#pragma unroll
    for (int mi = 0; mi < 2; ++mi) {
      int m = (mt0 + mi) * 16 + (lane & 15);
      const v8h* pa = (const v8h*)(Al + m * 32 + ((lane >> 4) << 3));
      v16h a = mk16(pa[0], pa[2]);
#pragma unroll
      for (int nt = 0; nt < 4; ++nt) acc[mi][nt] = wmma16(a, bfr[nt], acc[mi][nt]);
    }
    __syncthreads();
  }

  // spill C tiles to LDS for the cross-row IF scan
#pragma unroll
  for (int mi = 0; mi < 2; ++mi)
#pragma unroll
    for (int nt = 0; nt < 4; ++nt)
#pragma unroll
      for (int r = 0; r < 8; ++r) {
        int row = (mt0 + mi) * 16 + r + ((lane >> 4) << 3);
        int f = nt * 16 + (lane & 15);
        hb[row * 64 + f] = acc[mi][nt][r];
      }
  __syncthreads();

  // IF over t (row = t*32 + n), write spikes fp16
  for (int i = tid; i < 2048; i += 256) {
    int nn = i >> 6, f = i & 63;
    float vv = 0.f;
#pragma unroll
    for (int t = 0; t < TT; ++t) {
      vv += hb[(t * 32 + nn) * 64 + f];
      float s = (vv >= 1.f) ? 1.f : 0.f;
      s3[(size_t)(t * NB + nn) * 4096 + fb + f] = (_Float16)s;
      vv = (s > 0.f) ? 0.f : vv;
    }
  }
}

// ---------------------------------------------------------------------------
// Kernel 4: fc2 (4096->10) + IF + time-mean. One block per batch sample.
// ---------------------------------------------------------------------------
__global__ __launch_bounds__(256) void k_fc2(const _Float16* __restrict__ s3,
                                             const float* __restrict__ fc2w,
                                             float* __restrict__ out) {
  __shared__ float red[80 * 256];
  __shared__ float tot[80];
  const int n = blockIdx.x, tid = threadIdx.x;

  float part[80];
#pragma unroll
  for (int j = 0; j < 80; ++j) part[j] = 0.f;

  for (int k = tid; k < 4096; k += 256) {
    float sv[TT];
#pragma unroll
    for (int t = 0; t < TT; ++t)
      sv[t] = (float)s3[(size_t)(t * NB + n) * 4096 + k];
#pragma unroll
    for (int o = 0; o < 10; ++o) {
      float w = fc2w[o * 4096 + k];
#pragma unroll
      for (int t = 0; t < TT; ++t)
        part[t * 10 + o] = fmaf(sv[t], w, part[t * 10 + o]);
    }
  }
#pragma unroll
  for (int j = 0; j < 80; ++j) red[j * 256 + tid] = part[j];
  __syncthreads();
  if (tid < 80) {
    float s = 0.f;
    for (int i = 0; i < 256; ++i) s += red[tid * 256 + i];
    tot[tid] = s;
  }
  __syncthreads();
  if (tid < 10) {
    float vv = 0.f, accf = 0.f;
#pragma unroll
    for (int t = 0; t < TT; ++t) {
      vv += tot[t * 10 + tid];
      float s = (vv >= 1.f) ? 1.f : 0.f;
      accf += s;
      vv = (s > 0.f) ? 0.f : vv;
    }
    out[n * 10 + tid] = accf * 0.125f;
  }
}

// ---------------------------------------------------------------------------
extern "C" void kernel_launch(void* const* d_in, const int* in_sizes, int n_in,
                              void* d_out, int out_size, void* d_ws,
                              size_t ws_size, hipStream_t stream) {
  const float* x    = (const float*)d_in[0];
  const float* w1   = (const float*)d_in[1];
  const float* w2   = (const float*)d_in[2];
  const float* fc1w = (const float*)d_in[3];
  const float* fc2w = (const float*)d_in[4];
  float* out = (float*)d_out;

  char* ws = (char*)d_ws;
  // ws layout (all 256B aligned):
  //   s1 : T*N*64*32*32 fp16 = 33,554,432 B
  //   s2 : T*N*16384    fp16 =  8,388,608 B
  //   s3 : T*N*4096     fp16 =  2,097,152 B
  //   w2h: 64*576       fp16 =     73,728 B         (total ~44.1 MB)
  _Float16* s1  = (_Float16*)(ws);
  _Float16* s2  = (_Float16*)(ws + 33554432u);
  _Float16* s3  = (_Float16*)(ws + 33554432u + 8388608u);
  _Float16* w2h = (_Float16*)(ws + 33554432u + 8388608u + 2097152u);

  k_w2h<<<144, 256, 0, stream>>>(w2, w2h);
  k_conv1<<<dim3(32, 8, 8), 256, 0, stream>>>(x, w1, s1);
  k_conv2<<<dim3(32, 4, 4), 256, 0, stream>>>(s1, w2h, s2);
  k_fc1<<<64, 256, 0, stream>>>(s2, fc1w, s3);
  k_fc2<<<32, 256, 0, stream>>>(s3, fc2w, out);
}
